// CTM_grid_82815559402221
// MI455X (gfx1250) — matrix operand order repaired and checked by reading, not compiled, not run on hardware
//
#include <hip/hip_runtime.h>
#include <math.h>
#include <stdint.h>

typedef __attribute__((ext_vector_type(16))) _Float16 v16h;
typedef __attribute__((ext_vector_type(8)))  _Float16 v8h;
typedef __attribute__((ext_vector_type(8)))  float    v8f;

// ---------------- problem constants (from setup_inputs) ----------------
static constexpr int B    = 4;
static constexpr int N    = 4096;
static constexpr int C    = 128;
static constexpr int DOUT = 256;
static constexpr int H    = 64, W = 64, HW = 64 * 64;
static constexpr int OH   = 32, OW = 32, OHW = 32 * 32;
static constexpr int M    = 1024;          // ceil(N*0.25)
static constexpr float EPS = 1e-6f;
static constexpr float INVSQRTC = 0.0625f; // 1/sqrt(256)

static constexpr int LDK  = 264;   // LDS stride (halves) for K=256 tiles, bank-skewed
static constexpr int LDK2 = 136;   // LDS stride (halves) for K=128 tiles

// ---------------- CDNA5 async global->LDS DMA (ASYNCcnt path) ----------------
// LDS byte address = low 32 bits of the generic pointer (AS3 offset);
// global address in a 64-bit VGPR pair, saddr = off (GV mode).
__device__ __forceinline__ void async_g2l_b128(void* lds_dst, const void* gsrc) {
  const unsigned l            = (unsigned)(uintptr_t)lds_dst;
  const unsigned long long g  = (unsigned long long)(uintptr_t)gsrc;
  asm volatile("global_load_async_to_lds_b128 %0, %1, off" :: "v"(l), "v"(g) : "memory");
}
__device__ __forceinline__ void wait_async0() {
  asm volatile("s_wait_asynccnt 0x0" ::: "memory");
}

// ---------------- WMMA fragment loaders (ISA 7.12.2 layouts) ----------------
// A (16x32 f16): lane row = lane&15, hsel = lane>>4.
__device__ __forceinline__ v16h load_fragA(const _Float16* rowp, int ks, int hsel) {
  const v8h lo = *(const v8h*)(rowp + ks * 32 + hsel * 8);
  const v8h hi = *(const v8h*)(rowp + ks * 32 + hsel * 8 + 16);
  return __builtin_shufflevector(lo, hi, 0,1,2,3,4,5,6,7,8,9,10,11,12,13,14,15);
}
// B (32x16 f16): lane col = lane&15; per lane K = ks*32 + 8*hsel + e (e<8), ks*32 + 16 + 8*hsel + (e-8) (e>=8)
__device__ __forceinline__ v16h load_fragB(const _Float16* colp, int ks, int hsel) {
  const v8h lo = *(const v8h*)(colp + ks * 32 + hsel * 8);
  const v8h hi = *(const v8h*)(colp + ks * 32 + hsel * 8 + 16);
  return __builtin_shufflevector(lo, hi, 0,1,2,3,4,5,6,7,8,9,10,11,12,13,14,15);
}
#define WMMA_F16(a, b, c) \
  __builtin_amdgcn_wmma_f32_16x16x32_f16(false, (a), false, (b), (short)0, (c), false, false)

// Compute a 64x64 distance tile into LDS. Block = 128 threads (4 waves).
__device__ __forceinline__ void dist_tile(const _Float16* smA, const _Float16* smB,
                                          const float* sqA, const float* sqB, float* smD) {
  const int wave = threadIdx.x >> 5;
  const int lane = threadIdx.x & 31;
  const int hsel = lane >> 4;
  const int l16  = lane & 15;
  const _Float16* arow = smA + (wave * 16 + l16) * LDK;
  for (int c4 = 0; c4 < 4; ++c4) {
    const _Float16* bcol = smB + (c4 * 16 + l16) * LDK;
    v8f acc = {};
#pragma unroll
    for (int ks = 0; ks < 8; ++ks)
      acc = WMMA_F16(load_fragA(arow, ks, hsel), load_fragB(bcol, ks, hsel), acc);
#pragma unroll
    for (int r = 0; r < 8; ++r) {
      const int i = wave * 16 + r + 8 * hsel;
      const int j = c4 * 16 + l16;
      const float d2 = sqA[i] + sqB[j] - 2.0f * acc[r];
      smD[i * 64 + j] = sqrtf(fmaxf(d2, 0.0f)) * INVSQRTC;
    }
  }
}

// ---------------- small helpers ----------------
__device__ __forceinline__ int grid_idx(float lx, float ly, int g) {
  lx = fminf(fmaxf(lx, -1.f), 1.f) * 0.5f + 0.5f;
  ly = fminf(fmaxf(ly, -1.f), 1.f) * 0.5f + 0.5f;
  int col = (int)rintf(lx * (float)(g - 1));
  int row = (int)rintf(ly * (float)(g - 1));
  col = ::min(::max(col, 0), g - 1);
  row = ::min(::max(row, 0), g - 1);
  return row * g + col;
}

__device__ __forceinline__ float block_sum256(float v, float* sred) {
  const int tid = threadIdx.x;
  sred[tid] = v; __syncthreads();
  for (int s = 128; s > 0; s >>= 1) {
    if (tid < s) sred[tid] += sred[tid + s];
    __syncthreads();
  }
  const float r = sred[0]; __syncthreads();
  return r;
}

// ---------------- utility kernels ----------------
__global__ void k_fill(float* p, float v, int n) {
  int i = blockIdx.x * 256 + threadIdx.x;
  if (i < n) p[i] = v;
}
__global__ void k_cast_h(const float* src, _Float16* dst, int n) {
  int i = blockIdx.x * 256 + threadIdx.x;
  if (i < n) dst[i] = (_Float16)src[i];
}
// conv_w (DOUT,C,3,3) -> wr [kh*3+kw][ic][oc]
__global__ void k_wreorder(const float* cw, float* wr) {
  int i = blockIdx.x * 256 + threadIdx.x;
  if (i >= DOUT * C * 9) return;
  int k  = i % 9;
  int ic = (i / 9) % C;
  int oc = i / (9 * C);
  wr[(k * C + ic) * DOUT + oc] = cw[i];
}

// ---------------- token2map ----------------
__global__ void k_t2m_scatter(const float* x, const float* loc, const int* idx_agg,
                              float* gsum, float* gcnt) {
  const int n0 = blockIdx.x, b = blockIdx.y, c = threadIdx.x; // block 128
  const int ia = idx_agg[b * N + n0];
  const int hw = grid_idx(loc[(b * N + n0) * 2 + 0], loc[(b * N + n0) * 2 + 1], W);
  atomicAdd(&gsum[((size_t)b * HW + hw) * C + c], x[((size_t)b * N + ia) * C + c]);
  if (c == 0) atomicAdd(&gcnt[b * HW + hw], 1.0f);
}
__global__ void k_grid_div(float* gsum, const float* gcnt) {
  size_t i = (size_t)blockIdx.x * 256 + threadIdx.x;
  if (i >= (size_t)B * HW * C) return;
  gsum[i] = gsum[i] / (gcnt[i / C] + EPS);
}

// ---------------- conv 3x3 stride2 pad1, NHWC, 128->256 ----------------
__global__ void k_conv(const float* grid, const float* wr, const float* cb, float* out) {
  const int pix = blockIdx.x, b = blockIdx.y;
  const int oh = pix >> 5, ow = pix & 31;
  __shared__ float patch[9 * C];
  const int tid = threadIdx.x;
  for (int i = tid; i < 9 * C; i += 256) {
    const int k = i / C, ic = i % C;
    const int ih = oh * 2 - 1 + k / 3, iw = ow * 2 - 1 + k % 3;
    float v = 0.f;
    if (ih >= 0 && ih < H && iw >= 0 && iw < W)
      v = grid[(((size_t)b * H + ih) * W + iw) * C + ic];
    patch[i] = v;
  }
  __syncthreads();
  float acc = cb[tid];
  for (int k = 0; k < 9; ++k) {
#pragma unroll 8
    for (int ic = 0; ic < C; ++ic)
      acc = fmaf(patch[k * C + ic], wr[(k * C + ic) * DOUT + tid], acc);
  }
  out[((size_t)b * OHW + pix) * DOUT + tid] = acc;
}

// ---------------- skip GEMM: xt = x @ skip_w^T (WMMA, async LDS fill) ---------
__global__ void k_skip_wmma(const _Float16* xh, const _Float16* swh, float* xt) {
  const int nt = blockIdx.x, dt = blockIdx.y, b = blockIdx.z; // block 128
  __shared__ _Float16 smA[64 * LDK2];
  __shared__ _Float16 smB[64 * LDK2];
  const int tid = threadIdx.x;
  for (int i = tid; i < 64 * 16; i += 128) {
    const int row = i >> 4, kc = i & 15;
    async_g2l_b128(smA + row * LDK2 + kc * 8,
                   xh + ((size_t)b * N + nt * 64 + row) * C + kc * 8);
    async_g2l_b128(smB + row * LDK2 + kc * 8,
                   swh + (size_t)(dt * 64 + row) * C + kc * 8);
  }
  wait_async0();
  __syncthreads();
  const int wave = tid >> 5, lane = tid & 31, hsel = lane >> 4, l16 = lane & 15;
  const _Float16* arow = smA + (wave * 16 + l16) * LDK2;
  for (int c4 = 0; c4 < 4; ++c4) {
    const _Float16* bcol = smB + (c4 * 16 + l16) * LDK2;
    v8f acc = {};
#pragma unroll
    for (int ks = 0; ks < 4; ++ks)
      acc = WMMA_F16(load_fragA(arow, ks, hsel), load_fragB(bcol, ks, hsel), acc);
#pragma unroll
    for (int r = 0; r < 8; ++r) {
      const int row = nt * 64 + wave * 16 + r + 8 * hsel;
      const int col = dt * 64 + c4 * 16 + l16;
      xt[((size_t)b * N + row) * DOUT + col] = acc[r];
    }
  }
}

// ---------------- map2token ----------------
__global__ void k_m2t_tot(const int* idx_agg, const float* aw, float* tot) {
  int i = blockIdx.x * 256 + threadIdx.x;
  if (i >= B * N) return;
  atomicAdd(&tot[(i / N) * N + idx_agg[i]], aw[i]);
}
__global__ void k_m2t_scatter(const float* conv_out, const float* loc, const int* idx_agg,
                              const float* aw, const float* tot, float* xt) {
  const int n0 = blockIdx.x, b = blockIdx.y, c = threadIdx.x; // block 256
  const int ia = idx_agg[b * N + n0];
  const int hw = grid_idx(loc[(b * N + n0) * 2 + 0], loc[(b * N + n0) * 2 + 1], OW);
  const float coef = aw[b * N + n0] / (tot[b * N + ia] + EPS);
  atomicAdd(&xt[((size_t)b * N + ia) * DOUT + c],
            conv_out[((size_t)b * OHW + hw) * DOUT + c] * coef);
}

// ---------------- LayerNorm + conf + f16 recast + |x|^2 ----------------
__global__ void k_ln_conf(float* xt, _Float16* xth, float* sq, float* weight,
                          const float* g, const float* bb,
                          const float* cw, const float* cb) {
  const int n = blockIdx.x, b = blockIdx.y, c = threadIdx.x; // block 256
  __shared__ float sred[256];
  float* p = xt + ((size_t)b * N + n) * DOUT;
  const float v  = p[c];
  const float mu = block_sum256(v, sred) * (1.0f / DOUT);
  const float xc = v - mu;
  const float var = block_sum256(xc * xc, sred) * (1.0f / DOUT);
  const float y = xc * rsqrtf(var + 1e-5f) * g[c] + bb[c];
  p[c] = y;
  xth[((size_t)b * N + n) * DOUT + c] = (_Float16)y;
  const float s2 = block_sum256(y * y, sred);
  const float cf = block_sum256(y * cw[c], sred);
  if (c == 0) {
    sq[b * N + n]     = s2;
    weight[b * N + n] = expf(cf + cb[0]);
  }
}

// ---------------- DPC-KNN pass 1: density + global dist max ----------------
// Double-buffered B tiles: async DMA of tile ct+1 overlaps WMMA on tile ct.
__global__ void k_density(const _Float16* xth, const float* sq,
                          float* density, unsigned int* distmax) {
  const int rt = blockIdx.x, b = blockIdx.y;  // block 128
  __shared__ _Float16 smA[64 * LDK];
  __shared__ _Float16 smB[2][64 * LDK];
  __shared__ float smD[64 * 64];
  __shared__ float sqA[64], sqB[2][64];
  const int tid = threadIdx.x;
  const size_t base = (size_t)b * N;
  for (int i = tid; i < 64 * 32; i += 128) {
    const int row = i >> 5, kc = i & 31;
    async_g2l_b128(smA + row * LDK + kc * 8,
                   xth + (base + rt * 64 + row) * DOUT + kc * 8);
    async_g2l_b128(smB[0] + row * LDK + kc * 8,
                   xth + (base + row) * DOUT + kc * 8);
  }
  if (tid < 64) { sqA[tid] = sq[base + rt * 64 + tid]; sqB[0][tid] = sq[base + tid]; }
  wait_async0();
  __syncthreads();

  float t0 = 1e30f, t1 = 1e30f, t2 = 1e30f, t3 = 1e30f, t4 = 1e30f;
  float rmax = 0.0f;
  for (int ct = 0; ct < 64; ++ct) {
    const int cur = ct & 1, nxt = cur ^ 1;
    if (ct + 1 < 64) {                      // prefetch next tile (async DMA)
      for (int i = tid; i < 64 * 32; i += 128) {
        const int row = i >> 5, kc = i & 31;
        async_g2l_b128(smB[nxt] + row * LDK + kc * 8,
                       xth + (base + (ct + 1) * 64 + row) * DOUT + kc * 8);
      }
      if (tid < 64) sqB[nxt][tid] = sq[base + (ct + 1) * 64 + tid];
    }
    dist_tile(smA, smB[cur], sqA, sqB[cur], smD);
    __syncthreads();
    if (tid < 64) {
      const float* dr = smD + tid * 64;
      for (int j = 0; j < 64; ++j) {
        const float d = dr[j];
        rmax = fmaxf(rmax, d);
        if (d < t4) {
          t4 = d;
          if (t4 < t3) { float tt = t3; t3 = t4; t4 = tt; }
          if (t3 < t2) { float tt = t2; t2 = t3; t3 = tt; }
          if (t2 < t1) { float tt = t1; t1 = t2; t2 = tt; }
          if (t1 < t0) { float tt = t0; t0 = t1; t1 = tt; }
        }
      }
    }
    wait_async0();
    __syncthreads();
  }
  if (tid < 64) {
    const float ms = (t0*t0 + t1*t1 + t2*t2 + t3*t3 + t4*t4) * 0.2f;
    density[base + rt * 64 + tid] = expf(-ms);
    atomicMax(&distmax[b], __float_as_uint(rmax));
  }
}

// ---------------- DPC-KNN pass 2: parent distance -> score ----------------
__global__ void k_parent(const _Float16* xth, const float* sq, const float* density,
                         const unsigned int* distmax, float* score) {
  const int rt = blockIdx.x, b = blockIdx.y;
  __shared__ _Float16 smA[64 * LDK];
  __shared__ _Float16 smB[2][64 * LDK];
  __shared__ float smD[64 * 64];
  __shared__ float sqA[64], sqB[2][64], denB[2][64];
  const int tid = threadIdx.x;
  const size_t base = (size_t)b * N;
  for (int i = tid; i < 64 * 32; i += 128) {
    const int row = i >> 5, kc = i & 31;
    async_g2l_b128(smA + row * LDK + kc * 8,
                   xth + (base + rt * 64 + row) * DOUT + kc * 8);
    async_g2l_b128(smB[0] + row * LDK + kc * 8,
                   xth + (base + row) * DOUT + kc * 8);
  }
  if (tid < 64) {
    sqA[tid] = sq[base + rt * 64 + tid];
    sqB[0][tid] = sq[base + tid];
    denB[0][tid] = density[base + tid];
  }
  wait_async0();
  __syncthreads();

  const float dmax = __uint_as_float(distmax[b]);
  float runmin = dmax;
  const float myden = (tid < 64) ? density[base + rt * 64 + tid] : 0.0f;
  for (int ct = 0; ct < 64; ++ct) {
    const int cur = ct & 1, nxt = cur ^ 1;
    if (ct + 1 < 64) {
      for (int i = tid; i < 64 * 32; i += 128) {
        const int row = i >> 5, kc = i & 31;
        async_g2l_b128(smB[nxt] + row * LDK + kc * 8,
                       xth + (base + (ct + 1) * 64 + row) * DOUT + kc * 8);
      }
      if (tid < 64) {
        sqB[nxt][tid]  = sq[base + (ct + 1) * 64 + tid];
        denB[nxt][tid] = density[base + (ct + 1) * 64 + tid];
      }
    }
    dist_tile(smA, smB[cur], sqA, sqB[cur], smD);
    __syncthreads();
    if (tid < 64) {
      const float* dr = smD + tid * 64;
      for (int j = 0; j < 64; ++j)
        if (denB[cur][j] > myden) runmin = fminf(runmin, dr[j]);
    }
    wait_async0();
    __syncthreads();
  }
  if (tid < 64) score[base + rt * 64 + tid] = runmin * myden;
}

// ---------------- top-M selection (bitonic sort, 1 block / batch) ----------------
__global__ void k_topM(const float* score, int* index_down) {
  __shared__ float sval[N];
  __shared__ int   sidx[N];
  const int b = blockIdx.x, tid = threadIdx.x;  // block 1024
  for (int i = tid; i < N; i += 1024) { sval[i] = score[b * N + i]; sidx[i] = i; }
  __syncthreads();
  for (int k = 2; k <= N; k <<= 1) {
    for (int j = k >> 1; j > 0; j >>= 1) {
      for (int i = tid; i < N; i += 1024) {
        const int ixj = i ^ j;
        if (ixj > i) {
          const bool desc = ((i & k) == 0);
          const float a = sval[i], c = sval[ixj];
          if (desc ? (a < c) : (a > c)) {
            sval[i] = c; sval[ixj] = a;
            const int t = sidx[i]; sidx[i] = sidx[ixj]; sidx[ixj] = t;
          }
        }
      }
      __syncthreads();
    }
  }
  for (int i = tid; i < M; i += 1024) index_down[b * M + i] = sidx[i];
}

// ---------------- gather cluster centers ----------------
__global__ void k_gather_centers(const _Float16* xth, const float* sq, const int* index_down,
                                 _Float16* xch, float* sqc) {
  const int m = blockIdx.x, b = blockIdx.y, c = threadIdx.x; // block 256
  const int src = index_down[b * M + m];
  xch[((size_t)b * M + m) * DOUT + c] = xth[((size_t)b * N + src) * DOUT + c];
  if (c == 0) sqc[b * M + m] = sq[b * N + src];
}

// ---------------- assign tokens to nearest center (WMMA + fused argmin) ------
__global__ void k_assign(const _Float16* xth, const float* sq,
                         const _Float16* xch, const float* sqc, int* idx_cluster) {
  const int nt = blockIdx.x, b = blockIdx.y;  // block 128
  __shared__ _Float16 smT[64 * LDK];
  __shared__ _Float16 smC[2][16 * LDK];
  __shared__ float sqT[64], sqC[2][16];
  const int tid = threadIdx.x;
  const size_t tb = (size_t)b * N + nt * 64;
  for (int i = tid; i < 64 * 32; i += 128) {
    const int row = i >> 5, kc = i & 31;
    async_g2l_b128(smT + row * LDK + kc * 8, xth + (tb + row) * DOUT + kc * 8);
  }
  for (int i = tid; i < 16 * 32; i += 128) {
    const int row = i >> 5, kc = i & 31;
    async_g2l_b128(smC[0] + row * LDK + kc * 8,
                   xch + ((size_t)b * M + row) * DOUT + kc * 8);
  }
  if (tid < 64) sqT[tid] = sq[tb + tid];
  if (tid < 16) sqC[0][tid] = sqc[b * M + tid];
  wait_async0();
  __syncthreads();

  const int wave = tid >> 5, lane = tid & 31, hsel = lane >> 4, l16 = lane & 15;
  const _Float16* bcol = smT + (wave * 16 + l16) * LDK;
  float minv = 1e30f; int mini = 0;
  for (int mt = 0; mt < 64; ++mt) {
    const int cur = mt & 1, nxt = cur ^ 1;
    if (mt + 1 < 64) {
      for (int i = tid; i < 16 * 32; i += 128) {
        const int row = i >> 5, kc = i & 31;
        async_g2l_b128(smC[nxt] + row * LDK + kc * 8,
                       xch + ((size_t)b * M + (mt + 1) * 16 + row) * DOUT + kc * 8);
      }
      if (tid < 16) sqC[nxt][tid] = sqc[b * M + (mt + 1) * 16 + tid];
    }
    const _Float16* arow = smC[cur] + l16 * LDK;
    v8f acc = {};
#pragma unroll
    for (int ks = 0; ks < 8; ++ks)
      acc = WMMA_F16(load_fragA(arow, ks, hsel), load_fragB(bcol, ks, hsel), acc);
    const float st = sqT[wave * 16 + l16];
#pragma unroll
    for (int r = 0; r < 8; ++r) {
      const int ml = r + 8 * hsel;
      const float d2 = sqC[cur][ml] + st - 2.0f * acc[r];
      if (d2 < minv) { minv = d2; mini = mt * 16 + ml; }
    }
    wait_async0();
    __syncthreads();
  }
  const float ov = __shfl_xor(minv, 16, 32);
  const int   oi = __shfl_xor(mini, 16, 32);
  if (ov < minv) { minv = ov; mini = oi; }
  if (hsel == 0) idx_cluster[tb + wave * 16 + l16] = mini;
}
__global__ void k_set_centers(const int* index_down, int* idx_cluster) {
  int i = blockIdx.x * 256 + threadIdx.x;
  if (i >= B * M) return;
  const int b = i / M, m = i % M;
  idx_cluster[b * N + index_down[i]] = m;
}

// ---------------- merge ----------------
__global__ void k_allw(const int* idx_cluster, const float* weight, float* allw) {
  int i = blockIdx.x * 256 + threadIdx.x;
  if (i >= B * N) return;
  atomicAdd(&allw[(i / N) * M + idx_cluster[i]], weight[i]);
}
__global__ void k_xdown(const float* xt, const int* idx_cluster, const float* weight,
                        const float* allw, float* xdown, float* norm_w) {
  const int n = blockIdx.x, b = blockIdx.y, c = threadIdx.x; // block 256
  const int ic = idx_cluster[b * N + n];
  const float nw = weight[b * N + n] / (allw[b * M + ic] + EPS);
  atomicAdd(&xdown[((size_t)b * M + ic) * DOUT + c], xt[((size_t)b * N + n) * DOUT + c] * nw);
  if (c == 0) norm_w[b * N + n] = nw;
}
__global__ void k_down(const int* idx_agg, const int* idx_cluster, const float* norm_w,
                       const float* agg_w, float* out_idx, float* awd, unsigned int* maxw) {
  int i = blockIdx.x * 256 + threadIdx.x;
  if (i >= B * N) return;
  const int b = i / N;
  const int ia = idx_agg[i];
  out_idx[i] = (float)idx_cluster[b * N + ia];
  const float a = agg_w[i] * norm_w[b * N + ia];
  awd[i] = a;
  atomicMax(&maxw[b], __float_as_uint(a));
}
__global__ void k_awd_norm(const float* awd, const unsigned int* maxw, float* out_aw) {
  int i = blockIdx.x * 256 + threadIdx.x;
  if (i >= B * N) return;
  out_aw[i] = awd[i] / __uint_as_float(maxw[i / N]);
}

// ---------------- host side ----------------
static inline int cdiv(long long a, int b) { return (int)((a + b - 1) / b); }

extern "C" void kernel_launch(void* const* d_in, const int* in_sizes, int n_in,
                              void* d_out, int out_size, void* d_ws, size_t ws_size,
                              hipStream_t stream) {
  (void)in_sizes; (void)n_in; (void)out_size; (void)ws_size;
  const float* x        = (const float*)d_in[0];
  const float* loc_orig = (const float*)d_in[1];
  const int*   idx_agg  = (const int*)d_in[2];
  const float* agg_w    = (const float*)d_in[3];
  const float* conv_w   = (const float*)d_in[7];
  const float* conv_b   = (const float*)d_in[8];
  const float* skip_w   = (const float*)d_in[9];
  const float* ln_g     = (const float*)d_in[10];
  const float* ln_b     = (const float*)d_in[11];
  const float* conf_w   = (const float*)d_in[12];
  const float* conf_b   = (const float*)d_in[13];

  float* out_xdown = (float*)d_out;                    // B*M*DOUT
  float* out_idx   = out_xdown + (size_t)B * M * DOUT; // B*N
  float* out_aw    = out_idx + (size_t)B * N;          // B*N

  char* wsp = (char*)d_ws;
  auto carve = [&](size_t bytes) -> char* {
    char* p = wsp; wsp += (bytes + 255) & ~(size_t)255; return p;
  };
  float*     gsum     = (float*)carve((size_t)B * HW * C * 4);
  float*     gcnt     = (float*)carve((size_t)B * HW * 4);
  float*     conv_out = (float*)carve((size_t)B * OHW * DOUT * 4);
  _Float16*  xh       = (_Float16*)carve((size_t)B * N * C * 2);
  _Float16*  swh      = (_Float16*)carve((size_t)DOUT * C * 2);
  float*     wr       = (float*)carve((size_t)DOUT * C * 9 * 4);
  float*     xt       = (float*)carve((size_t)B * N * DOUT * 4);
  float*     tot      = (float*)carve((size_t)B * N * 4);
  _Float16*  xth      = (_Float16*)carve((size_t)B * N * DOUT * 2);
  float*     sq       = (float*)carve((size_t)B * N * 4);
  float*     weight   = (float*)carve((size_t)B * N * 4);
  float*     density  = (float*)carve((size_t)B * N * 4);
  unsigned*  distmax  = (unsigned*)carve((size_t)B * 4);
  float*     score    = (float*)carve((size_t)B * N * 4);
  int*       idx_down = (int*)carve((size_t)B * M * 4);
  _Float16*  xch      = (_Float16*)carve((size_t)B * M * DOUT * 2);
  float*     sqc      = (float*)carve((size_t)B * M * 4);
  int*       idx_clus = (int*)carve((size_t)B * N * 4);
  float*     allw     = (float*)carve((size_t)B * M * 4);
  float*     norm_w   = (float*)carve((size_t)B * N * 4);
  unsigned*  maxw     = (unsigned*)carve((size_t)B * 4);
  float*     awd      = (float*)carve((size_t)B * N * 4);

  // --- zero init ---
  k_fill<<<cdiv((size_t)B * HW * C, 256), 256, 0, stream>>>(gsum, 0.f, B * HW * C);
  k_fill<<<cdiv(B * HW, 256), 256, 0, stream>>>(gcnt, 0.f, B * HW);
  k_fill<<<cdiv(B * N, 256), 256, 0, stream>>>(tot, 0.f, B * N);
  k_fill<<<cdiv(B * M, 256), 256, 0, stream>>>(allw, 0.f, B * M);
  k_fill<<<1, 256, 0, stream>>>((float*)distmax, 0.f, B);
  k_fill<<<1, 256, 0, stream>>>((float*)maxw, 0.f, B);
  k_fill<<<cdiv((size_t)B * M * DOUT, 256), 256, 0, stream>>>(out_xdown, 0.f, B * M * DOUT);

  // --- precision casts / weight reorder ---
  k_cast_h<<<cdiv((size_t)B * N * C, 256), 256, 0, stream>>>(x, xh, B * N * C);
  k_cast_h<<<cdiv(DOUT * C, 256), 256, 0, stream>>>(skip_w, swh, DOUT * C);
  k_wreorder<<<cdiv(DOUT * C * 9, 256), 256, 0, stream>>>(conv_w, wr);

  // --- token2map + conv ---
  k_t2m_scatter<<<dim3(N, B), 128, 0, stream>>>(x, loc_orig, idx_agg, gsum, gcnt);
  k_grid_div<<<cdiv((size_t)B * HW * C, 256), 256, 0, stream>>>(gsum, gcnt);
  k_conv<<<dim3(OHW, B), DOUT, 0, stream>>>(gsum, wr, conv_b, conv_out);

  // --- skip GEMM (WMMA), then map2token scatter adds into xt ---
  k_skip_wmma<<<dim3(N / 64, DOUT / 64, B), 128, 0, stream>>>(xh, swh, xt);
  k_m2t_tot<<<cdiv(B * N, 256), 256, 0, stream>>>(idx_agg, agg_w, tot);
  k_m2t_scatter<<<dim3(N, B), DOUT, 0, stream>>>(conv_out, loc_orig, idx_agg, agg_w, tot, xt);

  // --- LN + conf + recast ---
  k_ln_conf<<<dim3(N, B), DOUT, 0, stream>>>(xt, xth, sq, weight, ln_g, ln_b, conf_w, conf_b);

  // --- DPC-KNN clustering (fused WMMA distance tiles, async double-buffer) ---
  k_density<<<dim3(N / 64, B), 128, 0, stream>>>(xth, sq, density, distmax);
  k_parent<<<dim3(N / 64, B), 128, 0, stream>>>(xth, sq, density, distmax, score);
  k_topM<<<B, 1024, 0, stream>>>(score, idx_down);
  k_gather_centers<<<dim3(M, B), DOUT, 0, stream>>>(xth, sq, idx_down, xch, sqc);
  k_assign<<<dim3(N / 64, B), 128, 0, stream>>>(xth, sq, xch, sqc, idx_clus);
  k_set_centers<<<cdiv(B * M, 256), 256, 0, stream>>>(idx_down, idx_clus);

  // --- weighted merge + downstream outputs ---
  k_allw<<<cdiv(B * N, 256), 256, 0, stream>>>(idx_clus, weight, allw);
  k_xdown<<<dim3(N, B), DOUT, 0, stream>>>(xt, idx_clus, weight, allw, out_xdown, norm_w);
  k_down<<<cdiv(B * N, 256), 256, 0, stream>>>(idx_agg, idx_clus, norm_w, agg_w,
                                               out_idx, awd, maxw);
  k_awd_norm<<<cdiv(B * N, 256), 256, 0, stream>>>(awd, maxw, out_aw);
}
